// HGCN_36386962932159
// MI455X (gfx1250) — compile-verified, matrix operand order checked by reading, not run on hardware
//
#include <hip/hip_runtime.h>
#include <hip/hip_fp16.h>
#include <math.h>

// ---------------------------------------------------------------------------
// Live computation: the hyperbolic branch is multiplied by an exact 0.0 in the
// reference, so only the second temporal-attention branch survives:
//   A[t][f] = sum_n U1_2[n] * x[n,t,f]                                (60 vals)
//   M[f][t] = sum_n U2_2[f][n] * (sum_f2 U3_2[f2]*x[n,t,f2])          (60 vals)
//   prod = A @ M ; s = sigmoid(prod + be_2) ; E = Ve_2 @ s
//   At2  = softmax(E, axis=1)                                         (12x12)
//   H[t'*5+fi][fo*12+t] = sum_kk conv2_w[fo,fi,kk]*At2[t', t-1+kk]    (60x60)
//   out[fo*96000 + n*12 + t] = conv2_b[fo] + sum_k x[n*60+k]*H[k][fo*12+t]
// Last step = (8000x60)@(60x60) f32 GEMM -> V_WMMA_F32_16X16X4_F32.
// H operand is DMA'd into LDS by the Tensor Data Mover with hardware row
// padding (64 dwords data + 16 dwords pad = stride 80, bank-conflict free).
// ---------------------------------------------------------------------------

#define NNODES 8000
#define TDIM   12
#define FDIM   5
#define KROW   60          // per-node contiguous vector length (t*5+f)
#define KPAD   64
#define XS_STRIDE 68       // LDS row stride for X tile (conflict-free A reads)
#define HS_STRIDE 80       // LDS row stride for H (disjoint banks low/high half)
#define ROWS_PER_BLK 128
#define WS_H_OFF 512       // float offset of H (64x64 row-major) in workspace

typedef __attribute__((ext_vector_type(2)))  float        v2f;
typedef __attribute__((ext_vector_type(8)))  float        v8f;
typedef __attribute__((ext_vector_type(16))) _Float16     v16h;
typedef __attribute__((ext_vector_type(4)))  unsigned int v4u;
typedef __attribute__((ext_vector_type(4)))  int          v4i;
typedef __attribute__((ext_vector_type(8)))  int          v8i;

// ---------------- kernel 0: zero the 128 accumulator floats ----------------
__global__ void hgcn_zero(float* __restrict__ acc) {
    acc[threadIdx.x] = 0.0f;
}

// ---------------- kernel 1: one-pass stats reductions over n ---------------
// 64 lanes per block; lane l<60 owns cell l of A (index l matches x row
// layout t*5+f) and cell l of M (f=l/12, t=l%12).
__global__ __launch_bounds__(64) void hgcn_reduce(
    const float* __restrict__ x,   // (N,T,F) contiguous, row n = 60 floats
    const float* __restrict__ U1,  // (N)
    const float* __restrict__ U2,  // (F,N)
    const float* __restrict__ U3,  // (F)
    float* __restrict__ acc)       // [0..59]=A, [64..123]=M
{
    const int l = threadIdx.x;
    const bool act = (l < KROW);
    const int fM = l / TDIM;
    const int tM = l - fM * TDIM;
    const float u3_0 = U3[0], u3_1 = U3[1], u3_2 = U3[2], u3_3 = U3[3], u3_4 = U3[4];
    float aA = 0.0f, aM = 0.0f;
    for (int n = blockIdx.x; n < NNODES; n += gridDim.x) {
        const float u1 = U1[n];                 // uniform -> scalar load
        const float* xr = x + n * KROW;
        __builtin_prefetch(xr + (int)gridDim.x * KROW, 0, 0);  // global_prefetch_b8
        if (act) {
            aA += u1 * xr[l];
            const float* xt = xr + tM * FDIM;
            float r = u3_0 * xt[0] + u3_1 * xt[1] + u3_2 * xt[2]
                    + u3_3 * xt[3] + u3_4 * xt[4];
            aM += U2[fM * NNODES + n] * r;
        }
    }
    if (act) {
        atomicAdd(&acc[l],      aA);            // global_atomic_add_f32
        atomicAdd(&acc[64 + l], aM);
    }
}

// ---------------- kernel 2: 12x12 attention chain + build H ----------------
__global__ __launch_bounds__(256) void hgcn_att(
    float* __restrict__ ws,         // [0..59]=A, [64..123]=M, [512..]=H out
    const float* __restrict__ be,   // (1,T,T)
    const float* __restrict__ Ve,   // (T,T)
    const float* __restrict__ w2)   // (F,F,1,3) conv2_w
{
    __shared__ float sS[TDIM * TDIM];
    __shared__ float sE[TDIM * TDIM];
    __shared__ float sAt[TDIM * TDIM];
    const int tid = threadIdx.x;

    if (tid < TDIM * TDIM) {        // prod + bias -> sigmoid
        const int t1 = tid / TDIM, t2 = tid - t1 * TDIM;
        float p = 0.0f;
        #pragma unroll
        for (int f = 0; f < FDIM; ++f)
            p += ws[t1 * FDIM + f] * ws[64 + f * TDIM + t2];
        p += be[tid];
        sS[tid] = 1.0f / (1.0f + __expf(-p));
    }
    __syncthreads();
    if (tid < TDIM * TDIM) {        // E = Ve @ s
        const int i = tid / TDIM, t2 = tid - i * TDIM;
        float e = 0.0f;
        #pragma unroll
        for (int j = 0; j < TDIM; ++j)
            e += Ve[i * TDIM + j] * sS[j * TDIM + t2];
        sE[tid] = e;
    }
    __syncthreads();
    if (tid < TDIM) {               // softmax over i, per column t2
        const int t2 = tid;
        float m = -1e30f;
        #pragma unroll
        for (int i = 0; i < TDIM; ++i) m = fmaxf(m, sE[i * TDIM + t2]);
        float sum = 0.0f;
        #pragma unroll
        for (int i = 0; i < TDIM; ++i) sum += __expf(sE[i * TDIM + t2] - m);
        const float inv = 1.0f / sum;
        #pragma unroll
        for (int i = 0; i < TDIM; ++i)
            sAt[i * TDIM + t2] = __expf(sE[i * TDIM + t2] - m) * inv;
    }
    __syncthreads();
    // H[k][j], k = t'*5+fi, j = fo*12+t   (zero padded to 64x64)
    for (int idx = tid; idx < KPAD * KPAD; idx += blockDim.x) {
        const int k = idx >> 6, j = idx & 63;
        float val = 0.0f;
        if (k < KROW && j < KROW) {
            const int tp = k / FDIM, fi = k - tp * FDIM;
            const int fo = j / TDIM, t  = j - fo * TDIM;
            #pragma unroll
            for (int kk = 0; kk < 3; ++kk) {
                const int tt = t - 1 + kk;
                if (tt >= 0 && tt < TDIM)
                    val += w2[fo * 15 + fi * 3 + kk] * sAt[tp * TDIM + tt];
            }
        }
        ws[WS_H_OFF + idx] = val;
    }
}

// ---------------- kernel 3: (8000x60)@(60x60) GEMM via f32 WMMA ------------
__global__ __launch_bounds__(256) void hgcn_gemm(
    const float* __restrict__ X,     // (N,60)
    const float* __restrict__ Hws,   // 64x64 row-major
    const float* __restrict__ bias,  // (F) conv2_b
    float* __restrict__ out)         // 480000 floats, y[fo][n][t]
{
    __shared__ __align__(16) float Xs[ROWS_PER_BLK * XS_STRIDE];
    __shared__ __align__(16) float Hs[KPAD * HS_STRIDE];
    const int tid = threadIdx.x;
    const int rowBase = blockIdx.x * ROWS_PER_BLK;

#if __has_builtin(__builtin_amdgcn_tensor_load_to_lds)
    // TDM: DMA the 64x64 dword H panel into LDS; hardware pad inserts 16
    // dwords after every 64 stored dwords -> LDS row stride 80 (= HS_STRIDE),
    // the same conflict-free layout the WMMA B-fragment reads expect.
    if (tid < 32) {                          // wave-uniform branch; one issue
        const unsigned long long ga = (unsigned long long)(uintptr_t)Hws;
        const unsigned int lds = (unsigned int)(uintptr_t)&Hs[0];  // addr[31:0] = LDS offset
        v4u g0;
        g0[0] = 1u;                          // count=1, user descriptor
        g0[1] = lds;                         // lds_addr (bytes)
        g0[2] = (unsigned int)ga;            // global_addr[31:0]
        g0[3] = ((unsigned int)(ga >> 32) & 0x01FFFFFFu) | (2u << 30);  // type=2
        v8i g1;
        g1[0] = (int)((2u << 16)             // data_size = 4B
                    | (1u << 20)             // pad_enable
                    | (5u << 22)             // pad_interval: 64 dwords
                    | (15u << 25));          // pad_amount: 16 dwords
        g1[1] = (int)(64u << 16);            // tensor_dim0 = 64
        g1[2] = (int)(64u << 16);            // tensor_dim1 = 64
        g1[3] = (int)(64u << 16);            // tile_dim0 = 64
        g1[4] = 64;                          // tile_dim1 = 64 (tile_dim2 = 0)
        g1[5] = 64;                          // tensor_dim0_stride = 64
        g1[6] = 0;
        g1[7] = 0;
        v4i gz = {0, 0, 0, 0};
    #if __clang_major__ >= 23
        v8i gz8 = {0, 0, 0, 0, 0, 0, 0, 0};
        __builtin_amdgcn_tensor_load_to_lds(g0, g1, gz, gz, gz8, 0);
    #else
        __builtin_amdgcn_tensor_load_to_lds(g0, g1, gz, gz, 0);
    #endif
        __builtin_amdgcn_s_wait_tensorcnt(0);   // data in LDS before barrier
    }
#else
    for (int i = tid; i < KPAD * KPAD; i += 256) {
        const int k = i >> 6, j = i & 63;
        Hs[k * HS_STRIDE + j] = Hws[i];
    }
#endif

    for (int i = tid; i < ROWS_PER_BLK * KPAD; i += 256) {
        const int r = i >> 6, c = i & 63;
        const int g = rowBase + r;
        float v = 0.0f;
        if (g < NNODES && c < KROW) v = X[g * KROW + c];
        Xs[r * XS_STRIDE + c] = v;
    }
    __syncthreads();

    const int wave  = tid >> 5;        // 0..7 -> 16-row tile
    const int lane  = tid & 31;
    const int m     = lane & 15;       // row within tile (A) / col within tile (B)
    const int lh    = lane >> 4;       // lane half selects K pair / row half
    const float* arow = &Xs[(wave * 16 + m) * XS_STRIDE];

    v8f acc[4] = {};                   // 4 col-tiles of 16x16 f32
#if __has_builtin(__builtin_amdgcn_wmma_f32_16x16x4_f32)
    #pragma unroll
    for (int kt = 0; kt < 16; ++kt) {
        const int kb = kt * 4 + lh * 2;            // ISA 16x4 f32 A layout
        v2f a; a.x = arow[kb]; a.y = arow[kb + 1];
        #pragma unroll
        for (int ct = 0; ct < 4; ++ct) {
            const int col = ct * 16 + m;
            v2f b;                                  // 4x16 f32 B layout (mirrored)
            b.x = Hs[kb * HS_STRIDE + col];
            b.y = Hs[(kb + 1) * HS_STRIDE + col];
            acc[ct] = __builtin_amdgcn_wmma_f32_16x16x4_f32(
                false, a, false, b, (short)0, acc[ct], false, false);
        }
    }
#else
    // Fallback: two K=32 steps of f16 WMMA (codegen-confirmed builtin).
    #pragma unroll
    for (int ks = 0; ks < 2; ++ks) {
        v16h a;
        #pragma unroll
        for (int e = 0; e < 16; ++e) {
            const int v = e >> 1;
            const int k = ks * 32 + 2 * (v & 3) + ((v >> 2) * 16) + (e & 1) + 8 * lh;
            a[e] = (_Float16)arow[k];
        }
        #pragma unroll
        for (int ct = 0; ct < 4; ++ct) {
            const int col = ct * 16 + m;
            v16h b;
            #pragma unroll
            for (int e = 0; e < 16; ++e) {
                const int v = e >> 1;
                const int k = ks * 32 + 2 * (v & 3) + ((v >> 2) * 16) + (e & 1) + 8 * lh;
                b[e] = (_Float16)Hs[k * HS_STRIDE + col];
            }
            acc[ct] = __builtin_amdgcn_wmma_f32_16x16x32_f16(
                false, a, false, b, (short)0, acc[ct], false, false);
        }
    }
#endif

    // C/D layout: VGPR r holds rows r (lanes 0-15) and 8+r (lanes 16-31).
    #pragma unroll
    for (int ct = 0; ct < 4; ++ct) {
        const int col = ct * 16 + m;
        if (col < KROW) {
            const int fo = col / TDIM, t = col - fo * TDIM;
            const float bv = bias[fo];
            #pragma unroll
            for (int r = 0; r < 8; ++r) {
                const int row = rowBase + wave * 16 + r + 8 * lh;
                if (row < NNODES)
                    out[fo * (NNODES * TDIM) + row * TDIM + t] = acc[ct][r] + bv;
            }
        }
    }
}

// ---------------------------------------------------------------------------
extern "C" void kernel_launch(void* const* d_in, const int* in_sizes, int n_in,
                              void* d_out, int out_size, void* d_ws, size_t ws_size,
                              hipStream_t stream) {
    const float* x    = (const float*)d_in[0];
    const float* U1_2 = (const float*)d_in[7];
    const float* U2_2 = (const float*)d_in[8];
    const float* U3_2 = (const float*)d_in[9];
    const float* be_2 = (const float*)d_in[10];
    const float* Ve_2 = (const float*)d_in[11];
    const float* c2w  = (const float*)d_in[14];
    const float* c2b  = (const float*)d_in[15];
    float* ws  = (float*)d_ws;
    float* out = (float*)d_out;

    hgcn_zero<<<1, 128, 0, stream>>>(ws);
    hgcn_reduce<<<32, 64, 0, stream>>>(x, U1_2, U2_2, U3_2, ws);
    hgcn_att<<<1, 256, 0, stream>>>(ws, be_2, Ve_2, c2w);
    const int nblk = (NNODES + ROWS_PER_BLK - 1) / ROWS_PER_BLK;  // 63
    hgcn_gemm<<<nblk, 256, 0, stream>>>(x, ws + WS_H_OFF, c2b, out);
}
